// Naive_DCN_34282428956880
// MI455X (gfx1250) — matrix operand order repaired and checked by reading, not compiled, not run on hardware
//
#include <hip/hip_runtime.h>
#include <hip/hip_bf16.h>

typedef __attribute__((ext_vector_type(16))) _Float16 v16h;
typedef __attribute__((ext_vector_type(8)))  _Float16 v8h;
typedef __attribute__((ext_vector_type(8)))  float    v8f;

#define KS      3
#define NTAP    9            // 3x3 taps
#define CIN     64
#define COUT    128
#define HH      128
#define WW      128
#define BB_     4
#define KDIM    (NTAP*CIN)   // 576 total K, ordering k = tap*64 + c
#define KSTEPS  (KDIM/32)    // 18 wmma k-steps
#define NCHUNK  3            // K chunks of 192 (3 taps x 64 ch)
#define KSPC    6            // k-steps per chunk
#define LDS_K   200          // padded f16 row stride (192 + 8) -> 400B, 16B aligned

static __device__ __forceinline__ int iclamp(int v, int lo, int hi) {
    return v < lo ? lo : (v > hi ? hi : v);
}

// ---------------------------------------------------------------------------
// Stage 1: offset-prediction conv (18ch, 3x3, pad 1), fp32, LDS-tiled.
// ---------------------------------------------------------------------------
__global__ __launch_bounds__(256) void dcn_offset_conv(
    const float* __restrict__ feat, const float* __restrict__ w_p,
    const float* __restrict__ b_p, float* __restrict__ offs)
{
    __shared__ float patch[18 * 18];
    const int tx = threadIdx.x, ty = threadIdx.y;
    const int x0 = blockIdx.x * 16, y0 = blockIdx.y * 16, bb = blockIdx.z;
    const int x = x0 + tx, y = y0 + ty;
    const int tid = ty * 16 + tx;
    const float* fb = feat + (size_t)bb * CIN * HH * WW;

    float acc[18];
#pragma unroll
    for (int i = 0; i < 18; ++i) acc[i] = 0.f;

    for (int c = 0; c < CIN; ++c) {
        // cooperative load of 18x18 zero-padded patch (origin y0-1, x0-1)
        for (int i = tid; i < 18 * 18; i += 256) {
            int py = y0 - 1 + i / 18;
            int px = x0 - 1 + i % 18;
            float v = 0.f;
            if (py >= 0 && py < HH && px >= 0 && px < WW)
                v = fb[c * HH * WW + py * WW + px];
            patch[i] = v;
        }
        __syncthreads();

        float v[9];
#pragma unroll
        for (int t = 0; t < 9; ++t)
            v[t] = patch[(ty + t / 3) * 18 + (tx + t % 3)];

        const float* wp = w_p + c * 9;   // w_p[oc][c][t] = wp[oc*CIN*9 + t]
#pragma unroll
        for (int t = 0; t < 9; ++t) {
            float vt = v[t];
#pragma unroll
            for (int oc = 0; oc < 18; ++oc)         // uniform addr -> s_load
                acc[oc] = fmaf(vt, wp[oc * CIN * 9 + t], acc[oc]);
        }
        __syncthreads();
    }
#pragma unroll
    for (int oc = 0; oc < 18; ++oc)
        offs[((size_t)(bb * 18 + oc) * HH + y) * WW + x] = acc[oc] + b_p[oc];
}

// ---------------------------------------------------------------------------
// Stage 2: pack w_conv (f32, [o][c][t]) into f16 WMMA B fragments.
// Fragment (ks, ntile): lane l holds 16 f16; lanes 0-15: K = ks*32+0..15,
// lanes 16-31: K = ks*32+16..31; VGPR j holds pair (2j, 2j+1); N = nt*16+(l&15).
// k ordering: k = tap*64 + c.
// ---------------------------------------------------------------------------
__global__ __launch_bounds__(256) void dcn_pack_w(
    const float* __restrict__ w_conv, _Float16* __restrict__ wB)
{
    const int id = blockIdx.x * 256 + threadIdx.x;
    if (id >= KSTEPS * 8 * 32) return;
    const int l  = id & 31;
    const int nt = (id >> 5) & 7;
    const int ks = id >> 8;
    const int n    = nt * 16 + (l & 15);
    const int koff = (l >> 4) * 16;

    v16h frag;
#pragma unroll
    for (int j = 0; j < 16; ++j) {
        int k = ks * 32 + koff + j;
        int t = k >> 6;          // tap
        int c = k & 63;          // channel
        frag[j] = (_Float16)w_conv[((size_t)n * CIN + c) * 9 + t];
    }
    *(v16h*)(wB + (size_t)id * 16) = frag;
}

// ---------------------------------------------------------------------------
// Stage 3: fused bilinear sampling + WMMA GEMM + coalesced NCHW store.
// One workgroup (8 wave32) per (b, y): M-tile = 128 pixels, N = 128 out-ch.
// ---------------------------------------------------------------------------
__global__ __launch_bounds__(256) void dcn_gemm(
    const float* __restrict__ feat, const float* __restrict__ offs,
    const _Float16* __restrict__ wB, const float* __restrict__ b_conv,
    float* __restrict__ out)
{
    __shared__ __align__(16) unsigned char smem_raw[128 * LDS_K * 2]; // 51200B
    _Float16* aLds = (_Float16*)smem_raw;
    float*    oLds = (float*)smem_raw;   // reused in epilogue (33792B)

    const int tid  = threadIdx.x;
    const int wg   = blockIdx.x;       // 0..511
    const int bb   = wg >> 7;          // batch
    const int y    = wg & 127;         // output row
    const int lane = tid & 31;
    const int nt   = tid >> 5;         // wave id == N-tile id

    const float* fb = feat + (size_t)bb * CIN * HH * WW;

    v8f cacc[8];
#pragma unroll
    for (int i = 0; i < 8; ++i) cacc[i] = v8f{};

    for (int ch = 0; ch < NCHUNK; ++ch) {
        __builtin_prefetch(wB + (((size_t)ch * KSPC * 8) + nt) * 32 * 16, 0, 1);

        // ---- sampling: fill aLds[128 px][192 k] in f16 ----
        const int npairs = (tid < 128) ? 2 : 1;  // 384 (px,tap) pairs / 256 thr
        for (int pp = 0; pp < npairs; ++pp) {
            const int pi = tid + pp * 256;
            const int px = pi & 127;     // x position
            const int tl = pi >> 7;      // local tap 0..2
            const int t  = ch * 3 + tl;  // global tap

            const float ox = offs[((size_t)(bb * 18 + t) * HH + y) * WW + px];
            const float oy = offs[((size_t)(bb * 18 + t + 9) * HH + y) * WW + px];
            // p = p0(+1 origin) + p_n + offset  (faithful to reference)
            const float pxf = (float)(y + 1 + (t / 3) - 1) + ox;
            const float pyf = (float)(px + 1 + (t % 3) - 1) + oy;
            const float flx = floorf(pxf), fly = floorf(pyf);
            const int ltx = iclamp((int)flx, 0, HH - 1);
            const int lty = iclamp((int)fly, 0, WW - 1);
            const int rbx = iclamp((int)flx + 1, 0, HH - 1);
            const int rby = iclamp((int)fly + 1, 0, WW - 1);
            const float pxc = fminf(fmaxf(pxf, 0.f), (float)(HH - 1));
            const float pyc = fminf(fmaxf(pyf, 0.f), (float)(WW - 1));
            const float wxl = 1.f + ((float)ltx - pxc);
            const float wyl = 1.f + ((float)lty - pyc);
            const float wxr = 1.f - ((float)rbx - pxc);
            const float wyr = 1.f - ((float)rby - pyc);
            const float glt = wxl * wyl, grb = wxr * wyr;
            const float glb = wxl * wyr, grt = wxr * wyl;
            const int ilt = ltx * WW + lty, irb = rbx * WW + rby;
            const int ilb = ltx * WW + rby, irt = rbx * WW + lty;

            _Float16* dst = aLds + px * LDS_K + tl * CIN;
            const float* Fp = fb;
#pragma unroll 4
            for (int c = 0; c < CIN; ++c) {
                float v = glt * Fp[ilt] + grb * Fp[irb]
                        + glb * Fp[ilb] + grt * Fp[irt];
                dst[c] = (_Float16)v;
                Fp += HH * WW;
            }
        }
        __syncthreads();

        // ---- WMMA: wave nt owns N-tile nt, sweeps 8 M-tiles ----
        for (int s = 0; s < KSPC; ++s) {
            const int ks = ch * KSPC + s;
            const _Float16* bp = wB + (((size_t)ks * 8 + nt) * 32 + lane) * 16;
            v16h bfrag = *(const v16h*)bp;                 // 2x global b128

            const int kb = s * 32 + 16 * (lane >> 4);     // A lane K base
            const _Float16* ap0 = aLds + (lane & 15) * LDS_K + kb;
#pragma unroll
            for (int mt = 0; mt < 8; ++mt) {
                const _Float16* ap = ap0 + mt * 16 * LDS_K;
                v8h lo = *(const v8h*)ap;                 // K k..k+7
                v8h hi = *(const v8h*)(ap + 8);           // K k+8..k+15
                v16h afrag = __builtin_shufflevector(lo, hi,
                    0,1,2,3,4,5,6,7,8,9,10,11,12,13,14,15);
                cacc[mt] = __builtin_amdgcn_wmma_f32_16x16x32_f16(
                    false, afrag, false, bfrag, (short)0, cacc[mt],
                    false, false);
            }
        }
        __syncthreads();
    }

    // ---- epilogue: C tiles -> LDS transpose -> coalesced NCHW stores ----
    const float bias = b_conv[nt * 16 + (lane & 15)];
    for (int half = 0; half < 2; ++half) {
        if ((nt >> 2) == half) {
            const int ol = (nt & 3) * 16 + (lane & 15);   // local out channel
#pragma unroll
            for (int mt = 0; mt < 8; ++mt) {
#pragma unroll
                for (int v = 0; v < 8; ++v) {
                    int x = mt * 16 + v + 8 * (lane >> 4);
                    oLds[ol * 132 + x] = cacc[mt][v] + bias;
                }
            }
        }
        __syncthreads();
        {
            const int r = tid >> 2;                       // 64 rows
            const int q = tid & 3;                        // 4 threads / row
            float* orow = out + (((size_t)bb * COUT + half * 64 + r) * HH + y) * WW;
            const float* lrow = oLds + r * 132;
#pragma unroll
            for (int i = 0; i < 8; ++i) {
                int x = (q * 8 + i) * 4;
                *(float4*)(orow + x) = *(const float4*)(lrow + x);
            }
        }
        __syncthreads();
    }
}

// ---------------------------------------------------------------------------
extern "C" void kernel_launch(void* const* d_in, const int* in_sizes, int n_in,
                              void* d_out, int out_size, void* d_ws, size_t ws_size,
                              hipStream_t stream) {
    const float* feat   = (const float*)d_in[0];
    const float* w_p    = (const float*)d_in[1];
    const float* b_p    = (const float*)d_in[2];
    const float* w_conv = (const float*)d_in[3];
    const float* b_conv = (const float*)d_in[4];
    float* out = (float*)d_out;

    char* ws = (char*)d_ws;
    _Float16* wB = (_Float16*)ws;                    // 18*8*32*16*2 = 147456 B
    float* offs  = (float*)(ws + 256 * 1024);        // 4*18*128*128*4 = 4.72 MB

    dcn_pack_w<<<(KSTEPS * 8 * 32 + 255) / 256, 256, 0, stream>>>(w_conv, wB);

    dim3 gb(WW / 16, HH / 16, BB_), tb(16, 16);
    dcn_offset_conv<<<gb, tb, 0, stream>>>(feat, w_p, b_p, offs);

    dcn_gemm<<<BB_ * HH, 256, 0, stream>>>(feat, offs, wB, b_conv, out);
}
